// FuseMoE_62405874811358
// MI455X (gfx1250) — matrix operand order repaired and checked by reading, not compiled, not run on hardware
//
#include <hip/hip_runtime.h>

typedef __attribute__((ext_vector_type(16))) __bf16 v16bf;
typedef __attribute__((ext_vector_type(8)))  __bf16 v8bf;
typedef __attribute__((ext_vector_type(8)))  float  v8f;
typedef __attribute__((ext_vector_type(4)))  float  v4f;

#define Bn 131072
#define Dn 256
#define Hn 256
#define En 4

// ---------------------------------------------------------------------------
// Workspace layout (d_ws):
//   [0       .. 524287 ]  packW1 : 262144 bf16, WMMA B-fragment layout
//   [524288  .. 1048575]  packW2 : 262144 bf16, WMMA B-fragment layout
//   [1048576 .. 1048591]  gcounts: 4 x uint32
//
// B-fragment packing (32x16 B matrix, wave32; contiguous k order, same as A, not AMD's table order):
//   lanes 0-15 hold K=0..15 (VGPR j = K 2j,2j+1), lanes 16-31 hold K=16..31,
//   N = lane & 15.  Packed so lane's 16 bf16 are contiguous:
//   elem(frag, lane, i) = W[e][kt*32 + (lane>>4)*16 + i][nt*16 + (lane&15)]
//   at offset frag*512 + lane*16 + i,   frag = (e*8 + kt)*16 + nt.
// ---------------------------------------------------------------------------

__global__ __launch_bounds__(256) void pack_weights(
    const float* __restrict__ W1, const float* __restrict__ W2,
    __bf16* __restrict__ p1, __bf16* __restrict__ p2)
{
    unsigned tid  = blockIdx.x * 256u + threadIdx.x;   // 0 .. 524287
    unsigned m    = tid >> 18;                         // 0 = W1, 1 = W2
    unsigned r    = tid & 262143u;
    unsigned frag = r >> 9;                            // 512 elems / frag
    unsigned li   = r & 511u;
    unsigned lane = li >> 4;
    unsigned i    = li & 15u;
    unsigned e    = frag >> 7;
    unsigned kt   = (frag >> 4) & 7u;
    unsigned nt   = frag & 15u;
    unsigned k    = kt * 32u + (lane >> 4) * 16u + i;
    unsigned n    = nt * 16u + (lane & 15u);
    const float* src = m ? W2 : W1;
    float v = src[(e * 256u + k) * 256u + n];
    (m ? p2 : p1)[r] = (__bf16)v;
}

__global__ __launch_bounds__(128) void moe_fused_kernel(
    const float* __restrict__ z,       const int* __restrict__ ridx_p,
    const float* __restrict__ cents,   const float* __restrict__ tau_raw,
    const float* __restrict__ bias1,   const float* __restrict__ bias2,
    const float* __restrict__ ln_g,    const float* __restrict__ ln_b,
    const __bf16* __restrict__ p1,     const __bf16* __restrict__ p2,
    float* __restrict__ out,           unsigned* __restrict__ gcounts)
{
    __shared__ float    c_lds[En * Dn];        // 4 KB  centroids (selected router)
    __shared__ float    wtab[4][16][En];       // 1 KB  per-row gate weight per expert
    __shared__ unsigned cnt_lds[En];
    __shared__ __bf16   h_lds[4][16 * Hn];     // 32 KB per-wave gelu(h) tile (bf16)

    const unsigned tid  = threadIdx.x;
    const unsigned wave = tid >> 5;
    const unsigned l    = tid & 31u;
    const unsigned half = l >> 4;              // which 16-lane half
    const unsigned lr   = l & 15u;             // row index (A) / N column (B,C,D)
    const unsigned tb   = (blockIdx.x * 4u + wave) * 16u;   // tile base row

    const int ridx = ridx_p[0];

    if (tid < En) cnt_lds[tid] = 0u;
    for (unsigned i = tid; i < En * Dn; i += 128u)
        c_lds[i] = cents[(unsigned)ridx * (En * Dn) + i];
    __syncthreads();

    float tr  = tau_raw[ridx];
    float tau = (tr > 20.f ? tr : log1pf(expf(tr))) + 1e-6f;

    // ---------------- gating: 2 lanes cooperate per row -------------------
    {
        float sq[En] = {0.f, 0.f, 0.f, 0.f};
        const float* zr = z + (size_t)(tb + lr) * Dn + half * 128u;
        const float* cb = c_lds + half * 128u;
        for (int i = 0; i < 128; ++i) {
            float zv = zr[i];
            #pragma unroll
            for (int e = 0; e < En; ++e) {
                float d = zv - cb[e * Dn + i];
                sq[e] += d * d;
            }
        }
        #pragma unroll
        for (int e = 0; e < En; ++e) sq[e] += __shfl_xor(sq[e], 16, 32);

        float s[En];
        #pragma unroll
        for (int e = 0; e < En; ++e) s[e] = expf(-sqrtf(sq[e]) / tau);

        int e0 = 0; float best0 = s[0];
        #pragma unroll
        for (int e = 1; e < En; ++e) if (s[e] > best0) { best0 = s[e]; e0 = e; }
        int e1 = -1; float best1 = -1e30f;
        #pragma unroll
        for (int e = 0; e < En; ++e) if (e != e0 && s[e] > best1) { best1 = s[e]; e1 = e; }

        float ex = expf(best1 - best0);            // softmax over [best0, best1]
        float w0 = 1.f / (1.f + ex);
        float w1 = ex  / (1.f + ex);

        if (half == 0) {
            #pragma unroll
            for (int e = 0; e < En; ++e)
                wtab[wave][lr][e] = (e == e0) ? w0 : ((e == e1) ? w1 : 0.f);
            atomicAdd(&cnt_lds[e0], 1u);
            atomicAdd(&cnt_lds[e1], 1u);
        }
    }
    __syncthreads();
    if (tid < En) atomicAdd(&gcounts[tid], cnt_lds[tid]);

    // -------- persistent A fragments: z tile (16x256) as bf16 -------------
    // A layout: lanes 0-15: K = 0..15; lanes 16-31: K = 16..31
    v16bf afrag[8];
    #pragma unroll
    for (int kt = 0; kt < 8; ++kt) {
        const float* zp = z + (size_t)(tb + lr) * Dn + kt * 32 + half * 16;
        v4f f0 = *(const v4f*)(zp);
        v4f f1 = *(const v4f*)(zp + 4);
        v4f f2 = *(const v4f*)(zp + 8);
        v4f f3 = *(const v4f*)(zp + 12);
        v16bf a;
        #pragma unroll
        for (int i = 0; i < 4; ++i) {
            a[i]      = (__bf16)f0[i];
            a[4 + i]  = (__bf16)f1[i];
            a[8 + i]  = (__bf16)f2[i];
            a[12 + i] = (__bf16)f3[i];
        }
        afrag[kt] = a;
    }

    // -------- moe accumulators, init = z (residual) + sum_e w_e*b2_e ------
    v8f moe[16];
    #pragma unroll
    for (int nt = 0; nt < 16; ++nt) {
        unsigned col = nt * 16u + lr;
        #pragma unroll
        for (int j = 0; j < 8; ++j) {
            unsigned M = half * 8u + j;
            float acc = z[(size_t)(tb + M) * Dn + col];
            #pragma unroll
            for (int e = 0; e < En; ++e)
                acc += wtab[wave][M][e] * bias2[e * Dn + col];
            moe[nt][j] = acc;
        }
    }

    // ---------------------------- expert loop -----------------------------
    for (int e = 0; e < En; ++e) {
        const __bf16* pw1 = p1 + (size_t)e * 128u * 512u;
        const __bf16* pw2 = p2 + (size_t)e * 128u * 512u;

        // GEMM-1: h = gelu(z @ W1 + b1), pre-scaled by gate weight
        for (int nt = 0; nt < 16; ++nt) {
            float bv = bias1[e * Hn + nt * 16 + lr];
            v8f acc;
            #pragma unroll
            for (int j = 0; j < 8; ++j) acc[j] = bv;
            #pragma unroll
            for (int kt = 0; kt < 8; ++kt) {
                v16bf bfr = *(const v16bf*)(pw1 + ((unsigned)(kt * 16 + nt)) * 512u + l * 16u);
                acc = __builtin_amdgcn_wmma_f32_16x16x32_bf16(
                          false, afrag[kt], false, bfr, (short)0, acc, false, false);
            }
            #pragma unroll
            for (int j = 0; j < 8; ++j) {
                unsigned M = half * 8u + j;
                float x = acc[j];
                float g = 0.5f * x * (1.f + erff(x * 0.70710678118654752f));  // exact gelu
                h_lds[wave][M * Hn + nt * 16 + lr] = (__bf16)(g * wtab[wave][M][e]);
            }
        }

        // GEMM-2: moe += (w_e * h_e) @ W2   (wave-local LDS; DS ops in-order)
        for (int kt = 0; kt < 8; ++kt) {
            const __bf16* hp = &h_lds[wave][lr * Hn + kt * 32 + half * 16];
            v8bf c0 = *(const v8bf*)(hp);
            v8bf c1 = *(const v8bf*)(hp + 8);
            v16bf ha = __builtin_shufflevector(c0, c1,
                        0,1,2,3,4,5,6,7,8,9,10,11,12,13,14,15);
            #pragma unroll
            for (int nt = 0; nt < 16; ++nt) {
                v16bf bfr = *(const v16bf*)(pw2 + ((unsigned)(kt * 16 + nt)) * 512u + l * 16u);
                moe[nt] = __builtin_amdgcn_wmma_f32_16x16x32_bf16(
                              false, ha, false, bfr, (short)0, moe[nt], false, false);
            }
        }
    }

    // ----------------------- LayerNorm epilogue ---------------------------
    const float inv = 1.f / 256.f;
    #pragma unroll
    for (int j = 0; j < 8; ++j) {
        float s1 = 0.f, s2 = 0.f;
        #pragma unroll
        for (int nt = 0; nt < 16; ++nt) {
            float x = moe[nt][j];
            s1 += x; s2 += x * x;
        }
        #pragma unroll
        for (int m = 1; m < 16; m <<= 1) {        // stays within 16-lane half
            s1 += __shfl_xor(s1, m, 32);
            s2 += __shfl_xor(s2, m, 32);
        }
        float mu   = s1 * inv;
        float var  = s2 * inv - mu * mu;
        float rstd = rsqrtf(var + 1e-5f);
        unsigned M = half * 8u + j;
        float* orow = out + (size_t)(tb + M) * Dn;
        #pragma unroll
        for (int nt = 0; nt < 16; ++nt) {
            unsigned col = nt * 16u + lr;
            orow[col] = ln_g[col] * (moe[nt][j] - mu) * rstd + ln_b[col];
        }
    }
}

__global__ void finalize_counts(const unsigned* __restrict__ gcounts,
                                float* __restrict__ out)
{
    if (threadIdx.x < En)
        out[(size_t)Bn * Dn + threadIdx.x] =
            (float)gcounts[threadIdx.x] * (1.f / 262144.f);   // B*K = 131072*2
}

extern "C" void kernel_launch(void* const* d_in, const int* in_sizes, int n_in,
                              void* d_out, int out_size, void* d_ws, size_t ws_size,
                              hipStream_t stream)
{
    const float* z    = (const float*)d_in[0];
    const int*   ridx = (const int*)  d_in[1];
    const float* cent = (const float*)d_in[2];
    const float* tau  = (const float*)d_in[3];
    const float* W1   = (const float*)d_in[4];
    const float* b1   = (const float*)d_in[5];
    const float* W2   = (const float*)d_in[6];
    const float* b2   = (const float*)d_in[7];
    const float* lng  = (const float*)d_in[8];
    const float* lnb  = (const float*)d_in[9];
    float* out = (float*)d_out;

    __bf16*   p1      = (__bf16*)d_ws;
    __bf16*   p2      = p1 + 262144;
    unsigned* gcounts = (unsigned*)((char*)d_ws + 1048576);

    hipMemsetAsync(gcounts, 0, En * sizeof(unsigned), stream);
    pack_weights<<<2048, 256, 0, stream>>>(W1, W2, p1, p2);
    moe_fused_kernel<<<2048, 128, 0, stream>>>(z, ridx, cent, tau, b1, b2,
                                               lng, lnb, p1, p2, out, gcounts);
    finalize_counts<<<1, 32, 0, stream>>>(gcounts, out);
}
